// nEq_NLMP3_18013092840060
// MI455X (gfx1250) — matrix operand order repaired and checked by reading, not compiled, not run on hardware
//
#include <hip/hip_runtime.h>

typedef __attribute__((ext_vector_type(16))) __bf16 v16bf;
typedef __attribute__((ext_vector_type(8)))  __bf16 v8bf;
typedef __attribute__((ext_vector_type(8)))  float  v8f;

#define CDIM 64
#define SHD  9
#define EBD  16
#define HID  256
#define XS_E 200   // LDS stride (halves) for edge X [64 x 192]
#define XS_N 136   // LDS stride (halves) for node X [64 x 128]
#define YSTR 264   // LDS stride (halves) for hidden Y [64 x 256]

// ---------------- fragment loaders (wave32 WMMA bf16 layouts) ----------------

__device__ __forceinline__ v16bf ld_afrag(const __bf16* base) {
  // base = &row[k0 + half*16]; A: elems 0..15 = K[k0+half*16 ..] contiguous
  v8bf lo = *(const v8bf*)(base);
  v8bf hi = *(const v8bf*)(base + 8);
  return __builtin_shufflevector(lo, hi, 0,1,2,3,4,5,6,7,8,9,10,11,12,13,14,15);
}

__device__ __forceinline__ v16bf ld_bfrag(const __bf16* base) {
  // base = &WT[n*K + k0 + half*16]; 16 contiguous halves
  v8bf lo = *(const v8bf*)(base);
  v8bf hi = *(const v8bf*)(base + 8);
  return __builtin_shufflevector(lo, hi, 0,1,2,3,4,5,6,7,8,9,10,11,12,13,14,15);
}

__device__ __forceinline__ v8f wmma_bf16(v16bf a, v16bf b, v8f c) {
  return __builtin_amdgcn_wmma_f32_16x16x32_bf16(false, a, false, b, (short)0, c,
                                                 false, false);
}

// GEMM: Y[64xHID] = relu(X[64xKK] * W^T + bias), Y stored bf16 in LDS.
// 8 waves: wave w owns N-tiles {2w, 2w+1}; loops over 4 M-tiles.
template<int KK, int XSTR>
__device__ __forceinline__ void mlp_hidden_gemm(const __bf16* Xls, const __bf16* WT,
                                                const float* bias, __bf16* Yls, int tid)
{
  const int lane = tid & 31, wave = tid >> 5;
  const int hf = lane >> 4, lm = lane & 15;
  const int nt0 = wave * 2, nt1 = nt0 + 1;
  const float bia0 = bias[nt0 * 16 + lm];
  const float bia1 = bias[nt1 * 16 + lm];
  const __bf16* bp0 = WT + (nt0 * 16 + lm) * KK + hf * 16;
  const __bf16* bp1 = WT + (nt1 * 16 + lm) * KK + hf * 16;
  for (int mt = 0; mt < 4; ++mt) {
    v8f acc0 = {}; v8f acc1 = {};
    const __bf16* arow = Xls + (mt * 16 + lm) * XSTR + hf * 16;
    #pragma unroll
    for (int ks = 0; ks < KK / 32; ++ks) {
      v16bf a  = ld_afrag(arow + ks * 32);
      v16bf b0 = ld_bfrag(bp0 + ks * 32);
      v16bf b1 = ld_bfrag(bp1 + ks * 32);
      acc0 = wmma_bf16(a, b0, acc0);
      acc1 = wmma_bf16(a, b1, acc1);
    }
    #pragma unroll
    for (int r = 0; r < 8; ++r) {
      float y0 = acc0[r] + bia0; y0 = y0 > 0.f ? y0 : 0.f;
      float y1 = acc1[r] + bia1; y1 = y1 > 0.f ? y1 : 0.f;
      int row = mt * 16 + hf * 8 + r;
      Yls[row * YSTR + nt0 * 16 + lm] = (__bf16)y0;
      Yls[row * YSTR + nt1 * 16 + lm] = (__bf16)y1;
    }
  }
}

// ---------------- weight prep: f32 -> bf16 transposed (+ padded) ----------------

__global__ __launch_bounds__(256) void prep_w(
    const float* wev1, const float* weu1, const float* weu2,
    const float* wnl1, const float* wnl2, const float* wev2, const float* wtp,
    __bf16* evt1, __bf16* eut1, __bf16* eut2, __bf16* nlt1, __bf16* nlt2,
    __bf16* evt2, __bf16* wtpT)
{
  int i = blockIdx.x * 256 + threadIdx.x;
  if (i < 49152) { int k = i / 256, n = i % 256; evt1[n * 192 + k] = (__bf16)wev1[i]; return; }
  i -= 49152;
  if (i < 49152) { int k = i / 256, n = i % 256; eut1[n * 192 + k] = (__bf16)weu1[i]; return; }
  i -= 49152;
  if (i < 16384) { int k = i / 64,  n = i % 64;  eut2[n * 256 + k] = (__bf16)weu2[i]; return; }
  i -= 16384;
  if (i < 32768) { int k = i / 256, n = i % 256; nlt1[n * 128 + k] = (__bf16)wnl1[i]; return; }
  i -= 32768;
  if (i < 16384) { int k = i / 64,  n = i % 64;  nlt2[n * 256 + k] = (__bf16)wnl2[i]; return; }
  i -= 16384;
  if (i < 4096) {  // evt2: [16 n][256 k], n>=9 zero-padded
    int n = i >> 8, k = i & 255;
    float v = (n < SHD) ? wev2[k * SHD + n] : 0.f;
    evt2[n * 256 + k] = (__bf16)v;
    return;
  }
  i -= 4096;
  if (i < 4096) {  // wtpT: [64 c][64 u]; U cols: 0..8=v, 16..24=fe, 25..40=fes
    int c = i >> 6, u = i & 63;
    int j = -1;
    if (u < SHD) j = u;
    else if (u >= 16 && u < 16 + SHD) j = SHD + (u - 16);
    else if (u >= 25 && u < 25 + EBD) j = 2 * SHD + (u - 25);
    float v = (j >= 0) ? wtp[j * 64 + c] : 0.f;
    wtpT[c * 64 + u] = (__bf16)v;
  }
}

__global__ __launch_bounds__(256) void zero_nf(float* nf, int n) {
  int i = blockIdx.x * 256 + threadIdx.x;
  if (i < n) nf[i] = 0.f;
}

// ---------------- edge kernel: 64 edges per block, 8 waves ----------------

__global__ __launch_bounds__(256) void edge_kernel(
    const float* __restrict__ hn, const float* __restrict__ he,
    const float* __restrict__ fe, const float* __restrict__ fes,
    const float* __restrict__ norm, const int* __restrict__ esrc,
    const int* __restrict__ edst,
    const __bf16* __restrict__ evt1, const float* __restrict__ b_ev1,
    const __bf16* __restrict__ evt2, const float* __restrict__ b_ev2,
    const __bf16* __restrict__ wtpT, const float* __restrict__ b_tp,
    const __bf16* __restrict__ eut1, const float* __restrict__ b_eu1,
    const __bf16* __restrict__ eut2, const float* __restrict__ b_eu2,
    float* __restrict__ out, float* __restrict__ nf, int NN, int Etot)
{
  __shared__ __bf16 Xs[64 * XS_E];
  __shared__ __bf16 Ys[64 * YSTR];

  const int tid = threadIdx.x;
  const int lane = tid & 31, wave = tid >> 5;
  const int hf = lane >> 4, lm = lane & 15;
  const int e0 = blockIdx.x * 64;

  // ---- stage 0: X = [he | hn[src] | hn[dst]]  (bf16) ----
  for (int idx = tid; idx < 64 * 192; idx += 256) {
    int r = idx / 192, c = idx % 192;
    int e = e0 + r;
    float val = 0.f;
    if (e < Etot) {
      if (c < 64)        val = he[(size_t)e * 64 + c];
      else if (c < 128)  val = hn[(size_t)esrc[e] * 64 + (c - 64)];
      else               val = hn[(size_t)edst[e] * 64 + (c - 128)];
    }
    Xs[r * XS_E + c] = (__bf16)val;
  }
  __syncthreads();

  // ---- stage 1: hidden1 = relu(X @ w_ev1 + b_ev1) ----
  mlp_hidden_gemm<192, XS_E>(Xs, evt1, b_ev1, Ys, tid);
  __syncthreads();

  // ---- stage 2: waves 0-3: v = hidden1 @ w_ev2 (WMMA, N padded to 16)
  //              -> U cols 0..15 (in-place in Xs[:,0:16]; he is dead).
  //      waves 4-7: stage fe|fes|zeros into U cols 16..63. ----
  if (wave < 4) {
    const int mt = wave;
    v8f acc = {};
    const __bf16* arow = Ys + (mt * 16 + lm) * YSTR + hf * 16;
    const __bf16* bp = evt2 + (size_t)lm * 256 + hf * 16;
    #pragma unroll
    for (int ks = 0; ks < 8; ++ks) {
      v16bf a = ld_afrag(arow + ks * 32);
      v16bf b = ld_bfrag(bp + ks * 32);
      acc = wmma_bf16(a, b, acc);
    }
    const float bia = (lm < SHD) ? b_ev2[lm] : 0.f;
    #pragma unroll
    for (int r = 0; r < 8; ++r) {
      int row = mt * 16 + hf * 8 + r;
      Xs[row * XS_E + lm] = (__bf16)(acc[r] + bia);
    }
  } else {
    for (int idx = tid - 128; idx < 64 * 48; idx += 128) {
      int r = idx / 48, cc = 16 + idx % 48;
      int e = e0 + r;
      float val = 0.f;
      if (e < Etot) {
        if (cc < 16 + SHD)            val = fe[(size_t)e * SHD + (cc - 16)];
        else if (cc >= 25 && cc < 41) val = fes[(size_t)e * EBD + (cc - 25)];
      }
      Xs[r * XS_E + cc] = (__bf16)val;
    }
  }
  __syncthreads();

  // ---- stage 3: t = U @ w_tp (WMMA, K padded to 64); t -> Xs[:,0:64] ----
  {
    const int mt  = wave & 3;
    const int ntA = (wave >> 2) * 2;
    v8f acc0 = {}; v8f acc1 = {};
    const __bf16* arow = Xs + (mt * 16 + lm) * XS_E + hf * 16;
    const __bf16* bp0 = wtpT + (size_t)(ntA * 16 + lm) * 64 + hf * 16;
    const __bf16* bp1 = wtpT + (size_t)((ntA + 1) * 16 + lm) * 64 + hf * 16;
    #pragma unroll
    for (int ks = 0; ks < 2; ++ks) {
      v16bf a  = ld_afrag(arow + ks * 32);
      v16bf b0 = ld_bfrag(bp0 + ks * 32);
      v16bf b1 = ld_bfrag(bp1 + ks * 32);
      acc0 = wmma_bf16(a, b0, acc0);
      acc1 = wmma_bf16(a, b1, acc1);
    }
    __syncthreads();   // all U reads complete before t overwrites U's columns
    const int c0 = ntA * 16 + lm, c1 = c0 + 16;
    const float bia0 = b_tp[c0], bia1 = b_tp[c1];
    #pragma unroll
    for (int r = 0; r < 8; ++r) {
      int row = mt * 16 + hf * 8 + r;
      Xs[row * XS_E + c0] = (__bf16)(acc0[r] + bia0);
      Xs[row * XS_E + c1] = (__bf16)(acc1[r] + bia1);
    }
  }
  __syncthreads();

  // ---- stage 4: hidden2 = relu([t|hs|hd] @ w_eu1 + b_eu1) ----
  mlp_hidden_gemm<192, XS_E>(Xs, eut1, b_eu1, Ys, tid);
  __syncthreads();

  // ---- stage 5: delta = hidden2 @ w_eu2; he_new = he + delta + b ----
  {
    const int mt  = wave & 3;
    const int ntA = (wave >> 2) * 2;
    v8f acc0 = {}; v8f acc1 = {};
    const __bf16* arow = Ys + (mt * 16 + lm) * YSTR + hf * 16;
    const __bf16* bp0 = eut2 + ((size_t)(ntA * 16 + lm)) * 256 + hf * 16;
    const __bf16* bp1 = eut2 + ((size_t)((ntA + 1) * 16 + lm)) * 256 + hf * 16;
    #pragma unroll
    for (int ks = 0; ks < 8; ++ks) {
      v16bf a  = ld_afrag(arow + ks * 32);
      v16bf b0 = ld_bfrag(bp0 + ks * 32);
      v16bf b1 = ld_bfrag(bp1 + ks * 32);
      acc0 = wmma_bf16(a, b0, acc0);
      acc1 = wmma_bf16(a, b1, acc1);
    }
    const int c0 = ntA * 16 + lm, c1 = c0 + 16;
    const float bia0 = b_eu2[c0], bia1 = b_eu2[c1];
    #pragma unroll
    for (int r = 0; r < 8; ++r) {
      int row = mt * 16 + hf * 8 + r;
      int e = e0 + row;
      if (e < Etot) {
        float nr = norm[e];
        int   d  = edst[e];
        float h0 = he[(size_t)e * 64 + c0] + acc0[r] + bia0;
        float h1 = he[(size_t)e * 64 + c1] + acc1[r] + bia1;
        out[(size_t)(NN + e) * 64 + c0] = h0;
        out[(size_t)(NN + e) * 64 + c1] = h1;
        atomicAdd(nf + (size_t)d * 64 + c0, h0 * nr);
        atomicAdd(nf + (size_t)d * 64 + c1, h1 * nr);
      }
    }
  }
}

// ---------------- node kernel: 64 nodes per block ----------------

__global__ __launch_bounds__(256) void node_kernel(
    const float* __restrict__ hn, const float* __restrict__ nf,
    const __bf16* __restrict__ nlt1, const float* __restrict__ b_nl1,
    const __bf16* __restrict__ nlt2, const float* __restrict__ b_nl2,
    float* __restrict__ out, int Ntot)
{
  __shared__ __bf16 Xs[64 * XS_N];
  __shared__ __bf16 Ys[64 * YSTR];

  const int tid = threadIdx.x;
  const int lane = tid & 31, wave = tid >> 5;
  const int hf = lane >> 4, lm = lane & 15;
  const int n0 = blockIdx.x * 64;

  // ---- stage 0: X = [hn | node_ftr]  (bf16) ----
  for (int idx = tid; idx < 64 * 128; idx += 256) {
    int r = idx >> 7, c = idx & 127;
    int node = n0 + r;
    float val = 0.f;
    if (node < Ntot)
      val = (c < 64) ? hn[(size_t)node * 64 + c] : nf[(size_t)node * 64 + (c - 64)];
    Xs[r * XS_N + c] = (__bf16)val;
  }
  __syncthreads();

  // ---- stage 1: hidden = relu(X @ w_nl1 + b_nl1) ----
  mlp_hidden_gemm<128, XS_N>(Xs, nlt1, b_nl1, Ys, tid);
  __syncthreads();

  // ---- stage 2: hn_new = hn + hidden @ w_nl2 + b_nl2 ----
  {
    const int mt  = wave & 3;
    const int ntA = (wave >> 2) * 2;
    v8f acc0 = {}; v8f acc1 = {};
    const __bf16* arow = Ys + (mt * 16 + lm) * YSTR + hf * 16;
    const __bf16* bp0 = nlt2 + ((size_t)(ntA * 16 + lm)) * 256 + hf * 16;
    const __bf16* bp1 = nlt2 + ((size_t)((ntA + 1) * 16 + lm)) * 256 + hf * 16;
    #pragma unroll
    for (int ks = 0; ks < 8; ++ks) {
      v16bf a  = ld_afrag(arow + ks * 32);
      v16bf b0 = ld_bfrag(bp0 + ks * 32);
      v16bf b1 = ld_bfrag(bp1 + ks * 32);
      acc0 = wmma_bf16(a, b0, acc0);
      acc1 = wmma_bf16(a, b1, acc1);
    }
    const int c0 = ntA * 16 + lm, c1 = c0 + 16;
    const float bia0 = b_nl2[c0], bia1 = b_nl2[c1];
    #pragma unroll
    for (int r = 0; r < 8; ++r) {
      int row = mt * 16 + hf * 8 + r;
      int node = n0 + row;
      if (node < Ntot) {
        out[(size_t)node * 64 + c0] = hn[(size_t)node * 64 + c0] + acc0[r] + bia0;
        out[(size_t)node * 64 + c1] = hn[(size_t)node * 64 + c1] + acc1[r] + bia1;
      }
    }
  }
}

// ---------------- launch ----------------

extern "C" void kernel_launch(void* const* d_in, const int* in_sizes, int n_in,
                              void* d_out, int out_size, void* d_ws, size_t ws_size,
                              hipStream_t stream)
{
  const float* hn   = (const float*)d_in[0];
  const float* he   = (const float*)d_in[1];
  const float* fe   = (const float*)d_in[2];
  const float* fes  = (const float*)d_in[3];
  const float* norm = (const float*)d_in[4];
  const int*   esrc = (const int*)d_in[5];
  const int*   edst = (const int*)d_in[6];
  const float* w_ev1 = (const float*)d_in[7];
  const float* b_ev1 = (const float*)d_in[8];
  const float* w_ev2 = (const float*)d_in[9];
  const float* b_ev2 = (const float*)d_in[10];
  const float* w_tp  = (const float*)d_in[11];
  const float* b_tp  = (const float*)d_in[12];
  const float* w_eu1 = (const float*)d_in[13];
  const float* b_eu1 = (const float*)d_in[14];
  const float* w_eu2 = (const float*)d_in[15];
  const float* b_eu2 = (const float*)d_in[16];
  const float* w_nl1 = (const float*)d_in[17];
  const float* b_nl1 = (const float*)d_in[18];
  const float* w_nl2 = (const float*)d_in[19];
  const float* b_nl2 = (const float*)d_in[20];

  const int Nn = in_sizes[0] / 64;   // 50,000
  const int Ee = in_sizes[1] / 64;   // 800,000

  float* out = (float*)d_out;
  char* ws = (char*)d_ws;
  float* nf = (float*)ws;
  size_t off = (size_t)Nn * 64 * 4;
  __bf16* evt1 = (__bf16*)(ws + off); off += 49152 * 2;
  __bf16* eut1 = (__bf16*)(ws + off); off += 49152 * 2;
  __bf16* eut2 = (__bf16*)(ws + off); off += 16384 * 2;
  __bf16* nlt1 = (__bf16*)(ws + off); off += 32768 * 2;
  __bf16* nlt2 = (__bf16*)(ws + off); off += 16384 * 2;
  __bf16* evt2 = (__bf16*)(ws + off); off += 4096 * 2;
  __bf16* wtpT = (__bf16*)(ws + off); off += 4096 * 2;

  const int nfElems = Nn * 64;
  zero_nf<<<(nfElems + 255) / 256, 256, 0, stream>>>(nf, nfElems);
  prep_w<<<(172032 + 255) / 256, 256, 0, stream>>>(
      w_ev1, w_eu1, w_eu2, w_nl1, w_nl2, w_ev2, w_tp,
      evt1, eut1, eut2, nlt1, nlt2, evt2, wtpT);

  const int eblocks = (Ee + 63) / 64;
  edge_kernel<<<eblocks, 256, 0, stream>>>(
      hn, he, fe, fes, norm, esrc, edst,
      evt1, b_ev1, evt2, b_ev2, wtpT, b_tp,
      eut1, b_eu1, eut2, b_eu2, out, nf, Nn, Ee);

  const int nblocks = (Nn + 63) / 64;
  node_kernel<<<nblocks, 256, 0, stream>>>(
      hn, nf, nlt1, b_nl1, nlt2, b_nl2, out, Nn);
}
